// BaseLayer_90881507983406
// MI455X (gfx1250) — matrix operand order repaired and checked by reading, not compiled, not run on hardware
//
#include <hip/hip_runtime.h>
#include <hip/hip_bf16.h>
#include <math.h>

// Problem constants (from reference): E=8 experts, D=1024, F=4096, T=B*S=8192 tokens.
#define E_NUM 8
#define D_DIM 1024
#define F_DIM 4096
#define T_TOK 8192
#define MT    64          // token rows per block tile (weight-reuse driver)

typedef _Float16 v16h __attribute__((ext_vector_type(16)));
typedef _Float16 v8h  __attribute__((ext_vector_type(8)));
typedef float    v8f  __attribute__((ext_vector_type(8)));
typedef unsigned int  u32x4 __attribute__((ext_vector_type(4)));
typedef int           i32x4 __attribute__((ext_vector_type(4)));
typedef int           i32x8 __attribute__((ext_vector_type(8)));

// ---------------------------------------------------------------------------
// fp32 -> fp16 weight pre-conversion (one pass; removes all cvt from GEMM hot
// loops and halves weight bytes so one expert's weights (16.8MB) live in L2).
// ---------------------------------------------------------------------------
__global__ __launch_bounds__(256)
void cvt_f32_to_f16(const float* __restrict__ src, _Float16* __restrict__ dst, int n) {
    const int i = (blockIdx.x * 256 + threadIdx.x) * 8;
    if (i >= n) return;
    const float* p = src + (size_t)i;
    v8h o;
#pragma unroll
    for (int j = 0; j < 8; ++j) o[j] = (_Float16)p[j];
    *(v8h*)(dst + (size_t)i) = o;
}

// ---------------------------------------------------------------------------
// Routing: per token 8 affinities, argmax (lowest index on ties),
// alpha = sigmoid(best affinity), bucket token under its expert.
// ---------------------------------------------------------------------------
__global__ __launch_bounds__(256)
void route_kernel(const float* __restrict__ x,
                  const float* __restrict__ cent,
                  int*   __restrict__ counts,
                  float* __restrict__ alphaOf,
                  int*   __restrict__ bucket) {
    __shared__ float cLds[E_NUM][D_DIM];
    for (int i = threadIdx.x; i < E_NUM * D_DIM; i += blockDim.x)
        (&cLds[0][0])[i] = cent[i];
    __syncthreads();

    const int wave = threadIdx.x >> 5;
    const int lane = threadIdx.x & 31;
    const int t = blockIdx.x * 8 + wave;
    if (t >= T_TOK) return;

    const float* xr = x + (size_t)t * D_DIM;
    float acc[E_NUM];
#pragma unroll
    for (int e = 0; e < E_NUM; ++e) acc[e] = 0.f;
    for (int k = lane; k < D_DIM; k += 32) {
        const float xv = xr[k];
#pragma unroll
        for (int e = 0; e < E_NUM; ++e) acc[e] += xv * cLds[e][k];
    }
#pragma unroll
    for (int e = 0; e < E_NUM; ++e)
#pragma unroll
        for (int off = 16; off > 0; off >>= 1)
            acc[e] += __shfl_xor(acc[e], off, 32);

    if (lane == 0) {
        int best = 0; float bv = acc[0];
#pragma unroll
        for (int e = 1; e < E_NUM; ++e)
            if (acc[e] > bv) { bv = acc[e]; best = e; }
        alphaOf[t] = 1.0f / (1.0f + expf(-bv));
        const int slot = atomicAdd(&counts[best], 1);
        bucket[best * T_TOK + slot] = t;
    }
}

// Tiny exclusive scan over 8 expert counts -> compacted H row offsets.
__global__ void scan_kernel(const int* __restrict__ counts, int* __restrict__ offsets) {
    if (threadIdx.x == 0) {
        int s = 0;
        for (int e = 0; e < E_NUM; ++e) { offsets[e] = s; s += counts[e]; }
        offsets[E_NUM] = s;
    }
}

// ---------------------------------------------------------------------------
// WMMA fragment helpers (wave32; A/B in this code's contiguous k order, C/D per ISA).
// A (16x32 f16): M = lane&15; K = ((lane&16)?16:0) + i (16 contiguous f16).
// B (32x16 f16): N = lane&15; K = ((lane&16)?16:0) + i (16 contiguous f16).
// C/D (16x16 f32): elem r -> M = r + ((lane&16)?8:0), N = lane&15.
// ---------------------------------------------------------------------------
__device__ __forceinline__ v16h load_a_frag_lds(const _Float16* __restrict__ row,
                                                int kk, int lane) {
    const int base = kk + ((lane & 16) ? 16 : 0);
    const v8h lo = *(const v8h*)(row + base);
    const v8h hi = *(const v8h*)(row + base + 8);
    v16h a;
#pragma unroll
    for (int i = 0; i < 8; ++i) { a[i] = lo[i]; a[i + 8] = hi[i]; }
    return a;
}

__device__ __forceinline__ v8f wmma_f16(v16h a, v16h b, v8f c) {
    return __builtin_amdgcn_wmma_f32_16x16x32_f16(false, a, false, b, (short)0, c,
                                                  false, false);
}

// ---------------------------------------------------------------------------
// Tensor Data Mover: 2D f16 tile load Global -> LDS (ISA ch.8 D# packing).
// Group0: [1:0]=count=1, [63:32]=lds_addr, [120:64]=global_addr, [127:126]=type=2
// Group1: data_size=1(2B), tensor_dim0/1, tile_dim0/1, tensor_dim0_stride
// 6-arg builtin on this toolchain: (g0, g1, g2, g3, g4, cpol).
// ---------------------------------------------------------------------------
__device__ __forceinline__ void tdm_load_2d_f16(unsigned ldsAddr,
                                                const _Float16* gptr,
                                                unsigned tensorDim0, unsigned tensorDim1,
                                                unsigned tileDim0,   unsigned tileDim1,
                                                unsigned stride0) {
    const unsigned long long ga = (unsigned long long)(size_t)gptr;
    u32x4 g0;
    g0[0] = 1u;                                   // count=1 (valid), no gather
    g0[1] = ldsAddr;                              // LDS byte address
    g0[2] = (unsigned)ga;                         // global_addr[31:0]
    g0[3] = ((unsigned)(ga >> 32) & 0x01FFFFFFu)  // global_addr[56:32]
          | 0x80000000u;                          // type=2 ("image")
    i32x8 g1;
    g1[0] = (int)(1u << 16);                      // data_size=1 -> 2 bytes/elem
    g1[1] = (int)((tensorDim0 & 0xFFFFu) << 16);  // tensor_dim0[15:0] @ bits[63:48]
    g1[2] = (int)((tensorDim0 >> 16) | ((tensorDim1 & 0xFFFFu) << 16));
    g1[3] = (int)((tensorDim1 >> 16) | ((tileDim0 & 0xFFFFu) << 16));
    g1[4] = (int)(tileDim1 & 0xFFFFu);            // tile_dim1; tile_dim2=0
    g1[5] = (int)stride0;                         // tensor_dim0_stride[31:0]
    g1[6] = 0;                                    // stride hi / dim1_stride = 0
    g1[7] = 0;
    i32x4 g2 = (i32x4){0, 0, 0, 0};
    i32x4 g3 = (i32x4){0, 0, 0, 0};
    i32x8 g4 = (i32x8){0, 0, 0, 0, 0, 0, 0, 0};
    __builtin_amdgcn_tensor_load_to_lds(g0, g1, g2, g3, g4, 0);
}

// ---------------------------------------------------------------------------
// G1: H[p, :] = relu( LN_e(x[tok]) @ w1[e]^T + b1[e] )  (f16 out, compacted p)
// Block = 64-token tile of one expert; w1[e] (f16) read exactly once per block.
// grid = (T/MT, E), block = 256 (8 waves).
// ---------------------------------------------------------------------------
__global__ __launch_bounds__(256)
void g1_kernel(const float* __restrict__ x,
               const float* __restrict__ ln_g,
               const float* __restrict__ ln_b,
               const _Float16* __restrict__ w1h,
               const float* __restrict__ b1,
               const int*   __restrict__ counts,
               const int*   __restrict__ offsets,
               const int*   __restrict__ bucket,
               _Float16* __restrict__ H) {
    __shared__ _Float16 lnA[MT][D_DIM];   // 128 KB
    __shared__ int toks[MT];

    const int e    = blockIdx.y;
    const int cnt  = counts[e];
    const int row0 = blockIdx.x * MT;
    if (row0 >= cnt) return;                       // uniform -> WMMA EXEC safe
    const int nrows = min(MT, cnt - row0);
    const int pbase = offsets[e] + row0;

    if (threadIdx.x < MT) {
        const int m = threadIdx.x;
        toks[m] = bucket[e * T_TOK + row0 + (m < nrows ? m : 0)];
    }
    __syncthreads();

    const int wave = threadIdx.x >> 5;
    const int lane = threadIdx.x & 31;

    // LayerNorm: 8 rows per wave.
    const float* g  = ln_g + (size_t)e * D_DIM;
    const float* bb = ln_b + (size_t)e * D_DIM;
#pragma unroll 1
    for (int rr = 0; rr < MT / 8; ++rr) {
        const int m = wave * (MT / 8) + rr;
        const float* xr = x + (size_t)toks[m] * D_DIM;
        float s = 0.f, s2 = 0.f;
        for (int k = lane; k < D_DIM; k += 32) { const float v = xr[k]; s += v; s2 += v * v; }
#pragma unroll
        for (int off = 16; off > 0; off >>= 1) {
            s += __shfl_xor(s, off, 32); s2 += __shfl_xor(s2, off, 32);
        }
        const float mu   = s  * (1.0f / D_DIM);
        const float var  = s2 * (1.0f / D_DIM) - mu * mu;
        const float rstd = rsqrtf(var + 1e-5f);
        for (int k = lane; k < D_DIM; k += 32)
            lnA[m][k] = (_Float16)((xr[k] - mu) * rstd * g[k] + bb[k]);
    }
    __syncthreads();

    // GEMM1: waves stripe the 256 N-tiles; 4 M-tiles each; K = 1024.
    const int kOff = (lane & 16) ? 16 : 0;
    const int moff = (lane & 16) ? 8 : 0;
#pragma unroll 1
    for (int nt = wave; nt < F_DIM / 16; nt += 8) {
        const int nGlob = nt * 16 + (lane & 15);
        const _Float16* wrow = w1h + ((size_t)e * F_DIM + nGlob) * D_DIM;
        v8f acc[4] = {};
#pragma unroll 2
        for (int kk = 0; kk < D_DIM; kk += 32) {
            const v16h b = *(const v16h*)(wrow + kk + kOff);   // f16: no cvt
#pragma unroll
            for (int mt = 0; mt < 4; ++mt) {
                const v16h a = load_a_frag_lds(&lnA[mt * 16 + (lane & 15)][0], kk, lane);
                acc[mt] = wmma_f16(a, b, acc[mt]);
            }
        }
        const float bias = b1[(size_t)e * F_DIM + nGlob];
#pragma unroll
        for (int mt = 0; mt < 4; ++mt)
#pragma unroll
            for (int r = 0; r < 8; ++r) {
                const int m = mt * 16 + moff + r;
                if (m < nrows) {
                    float v = acc[mt][r] + bias;
                    v = v > 0.f ? v : 0.f;
                    H[((size_t)(pbase + m)) * F_DIM + nGlob] = (_Float16)v;
                }
            }
    }
}

// ---------------------------------------------------------------------------
// G2: out[tok,:] = x[tok,:] + alpha * ( H[p,:] @ w2[e]^T + b2[e] )
// Block tile: 64 rows x 256 cols. H K-chunks (64x1024 f16 = 128KB) are
// DMA'd by the Tensor Data Mover into a ping-pong LDS pair (256KB) so the
// fetch of chunk c+1 overlaps the ~2048 WMMAs computing on chunk c.
// grid = (T/MT, E, D/256), block = 256 (8 waves).
// ---------------------------------------------------------------------------
__global__ __launch_bounds__(256)
void g2_kernel(const float* __restrict__ x,
               const _Float16* __restrict__ w2h,
               const float* __restrict__ b2,
               const int*   __restrict__ counts,
               const int*   __restrict__ offsets,
               const int*   __restrict__ bucket,
               const float* __restrict__ alphaOf,
               const _Float16* __restrict__ H,
               float* __restrict__ out) {
    __shared__ _Float16 hC[2][MT][1024];   // 256 KB ping-pong H chunks
    __shared__ int   toks[MT];
    __shared__ float alph[MT];

    const int e    = blockIdx.y;
    const int cnt  = counts[e];
    const int row0 = blockIdx.x * MT;
    if (row0 >= cnt) return;
    const int nrows = min(MT, cnt - row0);
    const int pbase = offsets[e] + row0;
    const int nbase = blockIdx.z * 256;

    if (threadIdx.x < MT) {
        const int m = threadIdx.x;
        const int t = bucket[e * T_TOK + row0 + (m < nrows ? m : 0)];
        toks[m] = t;
        alph[m] = alphaOf[t];
    }

    const int wave = threadIdx.x >> 5;
    const int lane = threadIdx.x & 31;
    const int kOff = (lane & 16) ? 16 : 0;
    const _Float16* Hbase = H + (size_t)pbase * F_DIM;
    const unsigned rowsLeft = (unsigned)(T_TOK - pbase);   // TDM OOB rows -> 0

    // Prologue: DMA chunk 0 into buffer 0 and wait for it.
    if (threadIdx.x < 32) {
        tdm_load_2d_f16((unsigned)(size_t)&hC[0][0][0], Hbase,
                        /*tensorDim0=*/F_DIM, /*tensorDim1=*/rowsLeft,
                        /*tileDim0=*/1024, /*tileDim1=*/MT, /*stride0=*/F_DIM);
        __builtin_amdgcn_s_wait_tensorcnt(0);
    }
    __syncthreads();

    v8f acc[8] = {};   // [nl=2][mt=4] output tiles per wave, persistent over K

#pragma unroll 1
    for (int chunk = 0; chunk < F_DIM / 1024; ++chunk) {
        const int cur = chunk & 1;

        // Kick off the next chunk's DMA into the other buffer (overlaps compute).
        if (chunk + 1 < F_DIM / 1024 && threadIdx.x < 32)
            tdm_load_2d_f16((unsigned)(size_t)&hC[1 - cur][0][0],
                            Hbase + (size_t)(chunk + 1) * 1024,
                            F_DIM, rowsLeft, 1024, MT, F_DIM);

        // Compute on the current buffer: 32 K-steps x 8 wmma.
#pragma unroll 2
        for (int kk = 0; kk < 1024; kk += 32) {
            v16h bfr[2];
#pragma unroll
            for (int nl = 0; nl < 2; ++nl) {
                const int nGlob = nbase + (wave * 2 + nl) * 16 + (lane & 15);
                bfr[nl] = *(const v16h*)(w2h + ((size_t)e * D_DIM + nGlob) * F_DIM
                                         + chunk * 1024 + kk + kOff);
            }
#pragma unroll
            for (int mt = 0; mt < 4; ++mt) {
                const v16h a = load_a_frag_lds(&hC[cur][mt * 16 + (lane & 15)][0],
                                               kk, lane);
#pragma unroll
                for (int nl = 0; nl < 2; ++nl)
                    acc[nl * 4 + mt] = wmma_f16(a, bfr[nl], acc[nl * 4 + mt]);
            }
        }

        __syncthreads();                       // all waves done reading hC[cur]
        if (threadIdx.x < 32)
            __builtin_amdgcn_s_wait_tensorcnt(0);   // next chunk landed
        __syncthreads();
    }

    // Epilogue: residual + alpha gate, valid rows only.
    const int moff = (lane & 16) ? 8 : 0;
#pragma unroll
    for (int nl = 0; nl < 2; ++nl) {
        const int col = nbase + (wave * 2 + nl) * 16 + (lane & 15);
        const float b2v = b2[(size_t)e * D_DIM + col];
#pragma unroll
        for (int mt = 0; mt < 4; ++mt)
#pragma unroll
            for (int r = 0; r < 8; ++r) {
                const int m = mt * 16 + moff + r;
                if (m < nrows) {
                    const int t = toks[m];
                    const float xv = x[(size_t)t * D_DIM + col];
                    out[(size_t)t * D_DIM + col] =
                        xv + alph[m] * (acc[nl * 4 + mt][r] + b2v);
                }
            }
    }
}

// ---------------------------------------------------------------------------
// Launcher (graph-capture safe). Workspace layout (assumes ws >= ~202 MB):
//   [0..32)        counts (8 int)
//   [64..100)      offsets (9 int)
//   [256..33024)   alphaOf (T f32)
//   [33024..)      bucket  (E*T int, 256KB)
//   [320K..)       w1h f16 (64MB), w2h f16 (64MB), H f16 (64MB)
// ---------------------------------------------------------------------------
extern "C" void kernel_launch(void* const* d_in, const int* in_sizes, int n_in,
                              void* d_out, int out_size, void* d_ws, size_t ws_size,
                              hipStream_t stream) {
    const float* x    = (const float*)d_in[0];
    const float* cent = (const float*)d_in[1];
    const float* ln_g = (const float*)d_in[2];
    const float* ln_b = (const float*)d_in[3];
    const float* w1   = (const float*)d_in[4];
    const float* b1   = (const float*)d_in[5];
    const float* w2   = (const float*)d_in[6];
    const float* b2   = (const float*)d_in[7];
    float* out = (float*)d_out;

    char* ws = (char*)d_ws;
    int*   counts  = (int*)(ws + 0);
    int*   offsets = (int*)(ws + 64);
    float* alphaOf = (float*)(ws + 256);
    int*   bucket  = (int*)(ws + 256 + (size_t)T_TOK * 4);
    const size_t EFD = (size_t)E_NUM * F_DIM * D_DIM;          // 33.55M elems
    _Float16* w1h = (_Float16*)(ws + 327680);
    _Float16* w2h = (_Float16*)(ws + 327680 + EFD * 2);
    _Float16* H   = (_Float16*)(ws + 327680 + EFD * 4);

    (void)hipMemsetAsync(counts, 0, E_NUM * sizeof(int), stream);

    const int nW = (int)EFD;
    cvt_f32_to_f16<<<(nW / 8 + 255) / 256, 256, 0, stream>>>(w1, w1h, nW);
    cvt_f32_to_f16<<<(nW / 8 + 255) / 256, 256, 0, stream>>>(w2, w2h, nW);

    route_kernel<<<T_TOK / 8, 256, 0, stream>>>(x, cent, counts, alphaOf, bucket);
    scan_kernel<<<1, 32, 0, stream>>>(counts, offsets);

    g1_kernel<<<dim3(T_TOK / MT, E_NUM), 256, 0, stream>>>(
        x, ln_g, ln_b, w1h, b1, counts, offsets, bucket, H);

    g2_kernel<<<dim3(T_TOK / MT, E_NUM, D_DIM / 256), 256, 0, stream>>>(
        x, w2h, b2, counts, offsets, bucket, alphaOf, H, out);
}
